// TMSA_57166014710353
// MI455X (gfx1250) — matrix operand order repaired and checked by reading, not compiled, not run on hardware
//
#include <hip/hip_runtime.h>
#include <hip/hip_bf16.h>

typedef __attribute__((ext_vector_type(16))) _Float16 v16h;
typedef __attribute__((ext_vector_type(8)))  _Float16 v8h;
typedef __attribute__((ext_vector_type(2)))  _Float16 v2h;
typedef __attribute__((ext_vector_type(8)))  float    v8f;

#define DIMC   120
#define HIDC   240
#define NH     6
#define HD     20
#define NTOK   256
#define NWIN   512
#define SCALE  0.22360679774997896f   // 1/sqrt(20)

// ---- workspace layout (bytes) ----
#define OFF_WQKV  0u          // 36*32*128 f16 = 294912
#define OFF_BQKV  294912u     // 36*32 f32     = 4608
#define OFF_WPROJ 299520u     // 128*256 f16   = 65536
#define OFF_BPROJ 365056u     // 128 f32       = 512
#define OFF_WFC11 365568u     // 240*128 f16   = 61440
#define OFF_BFC11 427008u     // 240 f32 (pad 1024)
#define OFF_WFC12 428032u     // 61440
#define OFF_BFC12 489472u     // 1024
#define OFF_WFC2  490496u     // 128*256 f16   = 65536
#define OFF_BFC2  556032u     // 512
#define OFF_XOUT  1048576u    // 512*256*256 f16 = 67108864 (fits in 192MB L2)

// ---------- WMMA helpers (CDNA5 16x16x32 f16 -> f32) ----------
__device__ __forceinline__ v16h mkfrag(const _Float16* lo, const _Float16* hi) {
  v8h a = *(const v8h*)lo;
  v8h b = *(const v8h*)hi;
  v16h r;
#pragma unroll
  for (int i = 0; i < 8; ++i) { r[i] = a[i]; r[i + 8] = b[i]; }
  return r;
}
// A-matrix 16x32 frag: lanes 0-15 row=l%16 K{0..15}; lanes 16-31 K{16..31}
__device__ __forceinline__ v16h ldA(const _Float16* row, int lhi) {
  const int k0 = lhi ? 16 : 0;
  return mkfrag(row + k0, row + k0 + 8);
}
// B-matrix 32x16 frag: lane=col, half-wave selects K group; 16 contiguous halves at p
__device__ __forceinline__ v16h ldB(const _Float16* p) {
  return mkfrag(p, p + 8);
}
__device__ __forceinline__ v8f wmma_f16(v16h a, v16h b, v8f c) {
  return __builtin_amdgcn_wmma_f32_16x16x32_f16(false, a, false, b, (short)0, c, false, false);
}

// window token -> flat token index in (8,4,64,64) grid
__device__ __forceinline__ int tok_flat(int w, int t) {
  int w0 = w & 7, h0 = (w >> 3) & 7, d0 = (w >> 6) & 1, n0 = w >> 7;
  int w1 = t & 7, h1 = (t >> 3) & 7, d1 = (t >> 6) & 1, n1 = t >> 7;
  int n = n0 * 2 + n1, d = d0 * 2 + d1;
  int hh = h0 * 8 + h1, ww = w0 * 8 + w1;
  return ((n * 4 + d) * 64 + hh) * 64 + ww;
}

// ================= kernel 0: weight repack to padded f16 =================
__global__ __launch_bounds__(256) void k_prep(
    const float* __restrict__ qsw, const float* __restrict__ qsb,
    const float* __restrict__ qmw, const float* __restrict__ qmb,
    const float* __restrict__ pw,  const float* __restrict__ pb,
    const float* __restrict__ f11w, const float* __restrict__ f11b,
    const float* __restrict__ f12w, const float* __restrict__ f12b,
    const float* __restrict__ f2w,  const float* __restrict__ f2b,
    char* __restrict__ ws)
{
  const int gid = blockIdx.x * blockDim.x + threadIdx.x;
  const int gsz = gridDim.x * blockDim.x;
  _Float16* Wqkv  = (_Float16*)(ws + OFF_WQKV);
  float*    bqkv  = (float*)(ws + OFF_BQKV);
  _Float16* Wproj = (_Float16*)(ws + OFF_WPROJ);
  float*    bproj = (float*)(ws + OFF_BPROJ);
  _Float16* Wf11  = (_Float16*)(ws + OFF_WFC11);
  float*    bf11  = (float*)(ws + OFF_BFC11);
  _Float16* Wf12  = (_Float16*)(ws + OFF_WFC12);
  float*    bf12  = (float*)(ws + OFF_BFC12);
  _Float16* Wf2   = (_Float16*)(ws + OFF_WFC2);
  float*    bf2   = (float*)(ws + OFF_BFC2);
  _Float16* xo    = (_Float16*)(ws + OFF_XOUT);

  // qkv weights: 36 groups [src(2) x which(3) x head(6)] of 32x128 (20x120 valid)
  for (int i = gid; i < 36 * 32 * 128; i += gsz) {
    int g = i >> 12, r = (i >> 7) & 31, c = i & 127;
    int src = g / 18, rem = g % 18, which = rem / 6, head = rem % 6;
    float v = 0.f;
    if (r < HD && c < DIMC) {
      const float* Wsrc = src ? qmw : qsw;
      v = Wsrc[(which * DIMC + head * HD + r) * DIMC + c];
    }
    Wqkv[i] = (_Float16)v;
  }
  for (int i = gid; i < 36 * 32; i += gsz) {
    int g = i >> 5, r = i & 31;
    int src = g / 18, rem = g % 18, which = rem / 6, head = rem % 6;
    float v = 0.f;
    if (r < HD) { const float* bs = src ? qmb : qsb; v = bs[which * DIMC + head * HD + r]; }
    bqkv[i] = v;
  }
  for (int i = gid; i < 128 * 256; i += gsz) {
    int r = i >> 8, c = i & 255;
    Wproj[i] = (_Float16)((r < DIMC && c < HIDC) ? pw[r * HIDC + c] : 0.f);
  }
  for (int i = gid; i < 128; i += gsz) bproj[i] = (i < DIMC) ? pb[i] : 0.f;
  for (int i = gid; i < 240 * 128; i += gsz) {
    int r = i >> 7, c = i & 127;
    Wf11[i] = (_Float16)((c < DIMC) ? f11w[r * DIMC + c] : 0.f);
  }
  for (int i = gid; i < 240; i += gsz) bf11[i] = f11b[i];
  for (int i = gid; i < 240 * 128; i += gsz) {
    int r = i >> 7, c = i & 127;
    Wf12[i] = (_Float16)((c < DIMC) ? f12w[r * DIMC + c] : 0.f);
  }
  for (int i = gid; i < 240; i += gsz) bf12[i] = f12b[i];
  for (int i = gid; i < 128 * 256; i += gsz) {
    int r = i >> 8, c = i & 255;
    Wf2[i] = (_Float16)((r < DIMC && c < HIDC) ? f2w[r * HIDC + c] : 0.f);
  }
  for (int i = gid; i < 128; i += gsz) bf2[i] = (i < DIMC) ? f2b[i] : 0.f;
  // zero the K-pad columns (240..255) of x_out rows so proj GEMM pads are clean
  for (int i = gid; i < NWIN * 256 * 16; i += gsz) {
    int row = i >> 4, c = 240 + (i & 15);
    xo[(size_t)row * 256 + c] = (_Float16)0.f;
  }
}

// ---- compile-time specialized QKV tile: straight-line stores, no branches ----
// TRANS=1: dst is [32][256] (transposed, packed b32 stores); ISQ: apply 1/sqrt(hd)
template <int TRANS, int ISQ>
__device__ __forceinline__ void qkv_tile(
    v16h a0, v16h a1, v16h a2, v16h a3,
    const _Float16* __restrict__ Wg, const float* __restrict__ bg,
    _Float16* __restrict__ dst, int mt, int lmod, int lhi)
{
#pragma unroll
  for (int nt = 0; nt < 2; ++nt) {
    const _Float16* wrow = Wg + (nt * 16 + lmod) * 128 + lhi * 16;
    v8f acc = {};
    acc = wmma_f16(a0, ldB(wrow), acc);
    acc = wmma_f16(a1, ldB(wrow + 32), acc);
    acc = wmma_f16(a2, ldB(wrow + 64), acc);
    acc = wmma_f16(a3, ldB(wrow + 96), acc);
    const int n = nt * 16 + lmod;
    const float bias = bg[n];
    if (TRANS) {
      _Float16* col = dst + n * 256 + mt * 16 + (lhi ? 8 : 0);
#pragma unroll
      for (int j = 0; j < 8; j += 2) {   // adjacent rows -> packed 32-bit store
        v2h pr;
        pr[0] = (_Float16)(acc[j] + bias);
        pr[1] = (_Float16)(acc[j + 1] + bias);
        *(v2h*)(col + j) = pr;
      }
    } else {
#pragma unroll
      for (int j = 0; j < 8; ++j) {
        int row = mt * 16 + (lhi ? j + 8 : j);
        float v = acc[j] + bias;
        if (ISQ) v *= SCALE;
        dst[row * 32 + n] = (_Float16)v;
      }
    }
  }
}

// ============ kernel 1: LN + QKV GEMM + self & mutual attention ============
// one block = one window (256 tokens), 8 waves.
// LDS: XN 64KB | qkv 96KB | per-wave P 64KB  (XF f32 staging aliases qkv+P)
__global__ __launch_bounds__(256) void k_qkv_attn(
    const float* __restrict__ x, const float* __restrict__ n1w,
    const float* __restrict__ n1b, const _Float16* __restrict__ Wqkv,
    const float* __restrict__ bqkv, _Float16* __restrict__ xout)
{
  extern __shared__ char smem[];
  _Float16* XN  = (_Float16*)smem;                 // [256][128]
  _Float16* QS  = (_Float16*)(smem + 65536);       // [256][32]
  _Float16* KSm = QS + 8192;                       // [256][32]
  _Float16* VST = QS + 16384;                      // [32][256] (transposed)
  _Float16* QMm = QS + 24576;                      // [256][32]
  _Float16* KMm = QS + 32768;                      // [256][32]
  _Float16* VMT = QS + 40960;                      // [32][256] (transposed)
  _Float16* P   = (_Float16*)(smem + 163840);      // [8 waves][16][256]
  float*    XF  = (float*)(smem + 65536);          // [256][120] staging (aliased)

  const int tid  = threadIdx.x;
  const int lane = tid & 31, wv = tid >> 5;
  const int lmod = lane & 15, lhi = lane >> 4;
  const int w = blockIdx.x;

  // gather window (coalesced-ish 480B runs)
  for (int i = tid; i < NTOK * DIMC; i += 256) {
    int t = i / DIMC, c = i - t * DIMC;
    XF[i] = x[(size_t)tok_flat(w, t) * DIMC + c];
  }
  __syncthreads();
  { // LayerNorm: one token per thread
    const float* xr = XF + tid * DIMC;
    float m = 0.f;
    for (int c = 0; c < DIMC; ++c) m += xr[c];
    m *= (1.0f / DIMC);
    float vv = 0.f;
    for (int c = 0; c < DIMC; ++c) { float d = xr[c] - m; vv += d * d; }
    vv *= (1.0f / DIMC);
    const float inv = rsqrtf(vv + 1e-5f);
    _Float16* nr = XN + tid * 128;
    for (int c = 0; c < DIMC; ++c) nr[c] = (_Float16)((xr[c] - m) * inv * n1w[c] + n1b[c]);
    for (int c = DIMC; c < 128; ++c) nr[c] = (_Float16)0.f;
  }
  __syncthreads();

  for (int h = 0; h < NH; ++h) {
    // prefetch this head's 6 weight groups (8KB each) into cache (global_prefetch_b8)
    if (tid < 64) {
#pragma unroll
      for (int m6 = 0; m6 < 6; ++m6) {
        const int g = (m6 < 3 ? 0 : 18) + (m6 % 3) * 6 + h;
        __builtin_prefetch((const char*)(Wqkv + (size_t)g * 4096) + tid * 128, 0, 3);
      }
    }
    // ---- per-head QKV GEMM: 256 x (6 mats x 32) x 128 ----
    for (int mi = 0; mi < 2; ++mi) {
      const int mt = wv + mi * 8;
      const _Float16* arow = XN + (mt * 16 + lmod) * 128;
      v16h a0 = ldA(arow, lhi), a1 = ldA(arow + 32, lhi);
      v16h a2 = ldA(arow + 64, lhi), a3 = ldA(arow + 96, lhi);
      const size_t GS = 32 * 128;
      qkv_tile<0, 1>(a0, a1, a2, a3, Wqkv + (size_t)(h)      * GS, bqkv + (h)      * 32, QS,  mt, lmod, lhi);
      qkv_tile<0, 0>(a0, a1, a2, a3, Wqkv + (size_t)(6 + h)  * GS, bqkv + (6 + h)  * 32, KSm, mt, lmod, lhi);
      qkv_tile<1, 0>(a0, a1, a2, a3, Wqkv + (size_t)(12 + h) * GS, bqkv + (12 + h) * 32, VST, mt, lmod, lhi);
      qkv_tile<0, 1>(a0, a1, a2, a3, Wqkv + (size_t)(18 + h) * GS, bqkv + (18 + h) * 32, QMm, mt, lmod, lhi);
      qkv_tile<0, 0>(a0, a1, a2, a3, Wqkv + (size_t)(24 + h) * GS, bqkv + (24 + h) * 32, KMm, mt, lmod, lhi);
      qkv_tile<1, 0>(a0, a1, a2, a3, Wqkv + (size_t)(30 + h) * GS, bqkv + (30 + h) * 32, VMT, mt, lmod, lhi);
    }
    __syncthreads();

    _Float16* pw = P + wv * 16 * 256;

    // ---- self attention: each wave owns M-tiles {wv, wv+8} ----
    for (int mi = 0; mi < 2; ++mi) {
      const int mt = wv + mi * 8;
      v16h aq = ldA(QS + (mt * 16 + lmod) * 32, lhi);
      v8f sacc[16];
#pragma unroll
      for (int nt = 0; nt < 16; ++nt) {
        v8f z = {};
        sacc[nt] = wmma_f16(aq, ldB(KSm + (nt * 16 + lmod) * 32 + lhi * 16), z);
      }
#pragma unroll
      for (int j = 0; j < 8; ++j) {           // row-wise softmax
        float mx = -3.0e38f;
#pragma unroll
        for (int t = 0; t < 16; ++t) mx = fmaxf(mx, sacc[t][j]);
#pragma unroll
        for (int m = 1; m < 16; m <<= 1) mx = fmaxf(mx, __shfl_xor(mx, m, 32));
        float s = 0.f;
#pragma unroll
        for (int t = 0; t < 16; ++t) { float e = __expf(sacc[t][j] - mx); sacc[t][j] = e; s += e; }
#pragma unroll
        for (int m = 1; m < 16; m <<= 1) s += __shfl_xor(s, m, 32);
        const float r = 1.0f / s;
        const int prow = lhi ? j + 8 : j;
#pragma unroll
        for (int t = 0; t < 16; ++t) pw[prow * 256 + t * 16 + lmod] = (_Float16)(sacc[t][j] * r);
      }
#pragma unroll
      for (int nt2 = 0; nt2 < 2; ++nt2) {     // O = P @ V
        v8f o = {};
#pragma unroll
        for (int kt = 0; kt < 8; ++kt) {
          v16h ap = ldA(pw + lmod * 256 + kt * 32, lhi);
          o = wmma_f16(ap, ldB(VST + (nt2 * 16 + lmod) * 256 + kt * 32 + lhi * 16), o);
        }
        const int n = nt2 * 16 + lmod;
        if (n < HD) {
#pragma unroll
          for (int j = 0; j < 8; ++j) {
            int row = mt * 16 + (lhi ? j + 8 : j);
            xout[(size_t)(w * 256 + row) * 256 + 120 + h * HD + n] = (_Float16)o[j];
          }
        }
      }
    }

    // ---- mutual attention: out tokens [0,128) use queries [128,256) and vice versa ----
    for (int half = 0; half < 2; ++half) {
      const int qb = half ? 0 : 128;
      const int kb = half ? 128 : 0;
      const int ob = half ? 128 : 0;
      const int mt = wv;                       // 8 M-tiles of 128 rows, one per wave
      v16h aq = ldA(QMm + (qb + mt * 16 + lmod) * 32, lhi);
      v8f macc[8];
#pragma unroll
      for (int nt = 0; nt < 8; ++nt) {
        v8f z = {};
        macc[nt] = wmma_f16(aq, ldB(KMm + (kb + nt * 16 + lmod) * 32 + lhi * 16), z);
      }
#pragma unroll
      for (int j = 0; j < 8; ++j) {
        float mx = -3.0e38f;
#pragma unroll
        for (int t = 0; t < 8; ++t) mx = fmaxf(mx, macc[t][j]);
#pragma unroll
        for (int m = 1; m < 16; m <<= 1) mx = fmaxf(mx, __shfl_xor(mx, m, 32));
        float s = 0.f;
#pragma unroll
        for (int t = 0; t < 8; ++t) { float e = __expf(macc[t][j] - mx); macc[t][j] = e; s += e; }
#pragma unroll
        for (int m = 1; m < 16; m <<= 1) s += __shfl_xor(s, m, 32);
        const float r = 1.0f / s;
        const int prow = lhi ? j + 8 : j;
#pragma unroll
        for (int t = 0; t < 8; ++t) pw[prow * 256 + t * 16 + lmod] = (_Float16)(macc[t][j] * r);
      }
#pragma unroll
      for (int nt2 = 0; nt2 < 2; ++nt2) {
        v8f o = {};
#pragma unroll
        for (int kt = 0; kt < 4; ++kt) {
          v16h ap = ldA(pw + lmod * 256 + kt * 32, lhi);
          o = wmma_f16(ap, ldB(VMT + (nt2 * 16 + lmod) * 256 + kb + kt * 32 + lhi * 16), o);
        }
        const int n = nt2 * 16 + lmod;
        if (n < HD) {
#pragma unroll
          for (int j = 0; j < 8; ++j) {
            int row = ob + mt * 16 + (lhi ? j + 8 : j);
            xout[(size_t)(w * 256 + row) * 256 + h * HD + n] = (_Float16)o[j];
          }
        }
      }
    }
    __syncthreads();
  }
}

// ============ kernel 2: proj(240->120) + window-reverse + residual ============
__global__ __launch_bounds__(256) void k_proj(
    const _Float16* __restrict__ xout, const _Float16* __restrict__ Wproj,
    const float* __restrict__ bproj, const float* __restrict__ xin,
    float* __restrict__ out)
{
  const int tid = threadIdx.x, lane = tid & 31, wv = tid >> 5;
  const int lmod = lane & 15, lhi = lane >> 4;
  const int w = blockIdx.x;
  for (int mi = 0; mi < 2; ++mi) {
    const int mt = wv + mi * 8;
    const _Float16* arow = xout + (size_t)(w * 256 + mt * 16 + lmod) * 256;
    v16h a[8];
#pragma unroll
    for (int kt = 0; kt < 8; ++kt) a[kt] = ldA(arow + kt * 32, lhi);
#pragma unroll
    for (int nt = 0; nt < 8; ++nt) {
      const _Float16* wrow = Wproj + (nt * 16 + lmod) * 256 + lhi * 16;
      v8f acc = {};
#pragma unroll
      for (int kt = 0; kt < 8; ++kt) acc = wmma_f16(a[kt], ldB(wrow + kt * 32), acc);
      const int o = nt * 16 + lmod;
      if (o < DIMC) {
        const float bias = bproj[o];
#pragma unroll
        for (int j = 0; j < 8; ++j) {
          int row = mt * 16 + (lhi ? j + 8 : j);
          size_t idx = (size_t)tok_flat(w, row) * DIMC + o;
          out[idx] = xin[idx] + acc[j] + bias;
        }
      }
    }
  }
}

// ============ kernel 3: LN2 + gated MLP + residual (in-place, token-local) ============
// LDS: H2 64KB | HID 128KB (f32 y1 staging aliases HID)
__global__ __launch_bounds__(256) void k_mlp(
    const float* __restrict__ n2w, const float* __restrict__ n2b,
    const _Float16* __restrict__ Wf11, const float* __restrict__ bf11,
    const _Float16* __restrict__ Wf12, const float* __restrict__ bf12,
    const _Float16* __restrict__ Wf2,  const float* __restrict__ bf2,
    float* __restrict__ out)
{
  extern __shared__ char smem[];
  _Float16* H2  = (_Float16*)smem;              // [256][128]
  _Float16* HID = (_Float16*)(smem + 65536);    // [256][256]
  float*    YF  = (float*)(smem + 65536);       // [256][120] staging (aliased)
  const int tid = threadIdx.x, lane = tid & 31, wv = tid >> 5;
  const int lmod = lane & 15, lhi = lane >> 4;
  const size_t tok0 = (size_t)blockIdx.x * 256;

  for (int i = tid; i < 256 * DIMC; i += 256) YF[i] = out[tok0 * DIMC + i];
  __syncthreads();
  { // LayerNorm2
    const float* xr = YF + tid * DIMC;
    float m = 0.f;
    for (int c = 0; c < DIMC; ++c) m += xr[c];
    m *= (1.0f / DIMC);
    float vv = 0.f;
    for (int c = 0; c < DIMC; ++c) { float d = xr[c] - m; vv += d * d; }
    vv *= (1.0f / DIMC);
    const float inv = rsqrtf(vv + 1e-5f);
    _Float16* nr = H2 + tid * 128;
    for (int c = 0; c < DIMC; ++c) nr[c] = (_Float16)((xr[c] - m) * inv * n2w[c] + n2b[c]);
    for (int c = DIMC; c < 128; ++c) nr[c] = (_Float16)0.f;
  }
  __syncthreads();
  for (int c = 240; c < 256; ++c) HID[tid * 256 + c] = (_Float16)0.f;  // K pad

  // GEMM1: paired fc11/fc12 tiles, exact gelu gate
  for (int mi = 0; mi < 2; ++mi) {
    const int mt = wv + mi * 8;
    const _Float16* arow = H2 + (mt * 16 + lmod) * 128;
    v16h a[4];
#pragma unroll
    for (int kt = 0; kt < 4; ++kt) a[kt] = ldA(arow + kt * 32, lhi);
    for (int p = 0; p < 15; ++p) {
      const _Float16* w1 = Wf11 + (p * 16 + lmod) * 128 + lhi * 16;
      const _Float16* w2 = Wf12 + (p * 16 + lmod) * 128 + lhi * 16;
      v8f acc1 = {}, acc2 = {};
#pragma unroll
      for (int kt = 0; kt < 4; ++kt) {
        acc1 = wmma_f16(a[kt], ldB(w1 + kt * 32), acc1);
        acc2 = wmma_f16(a[kt], ldB(w2 + kt * 32), acc2);
      }
      const int col = p * 16 + lmod;
      const float b1 = bf11[col], b2 = bf12[col];
#pragma unroll
      for (int j = 0; j < 8; ++j) {
        int row = mt * 16 + (lhi ? j + 8 : j);
        float g = acc1[j] + b1;
        g = 0.5f * g * (1.0f + erff(g * 0.70710678118f));
        HID[row * 256 + col] = (_Float16)(g * (acc2[j] + b2));
      }
    }
  }
  __syncthreads();
  // GEMM2: fc2 (240->120) + residual, in-place (each element read+written by one lane)
  for (int mi = 0; mi < 2; ++mi) {
    const int mt = wv + mi * 8;
    const _Float16* arow = HID + (mt * 16 + lmod) * 256;
    v16h a[8];
#pragma unroll
    for (int kt = 0; kt < 8; ++kt) a[kt] = ldA(arow + kt * 32, lhi);
#pragma unroll
    for (int nt = 0; nt < 8; ++nt) {
      const _Float16* wrow = Wf2 + (nt * 16 + lmod) * 256 + lhi * 16;
      v8f acc = {};
#pragma unroll
      for (int kt = 0; kt < 8; ++kt) acc = wmma_f16(a[kt], ldB(wrow + kt * 32), acc);
      const int col = nt * 16 + lmod;
      if (col < DIMC) {
        const float bias = bf2[col];
#pragma unroll
        for (int j = 0; j < 8; ++j) {
          int row = mt * 16 + (lhi ? j + 8 : j);
          size_t idx = (tok0 + row) * DIMC + col;
          out[idx] = out[idx] + acc[j] + bias;
        }
      }
    }
  }
}

extern "C" void kernel_launch(void* const* d_in, const int* in_sizes, int n_in,
                              void* d_out, int out_size, void* d_ws, size_t ws_size,
                              hipStream_t stream) {
  (void)in_sizes; (void)n_in; (void)out_size; (void)ws_size;
  const float* x    = (const float*)d_in[0];
  const float* n1w  = (const float*)d_in[1];
  const float* n1b  = (const float*)d_in[2];
  const float* qsw  = (const float*)d_in[3];
  const float* qsb  = (const float*)d_in[4];
  const float* qmw  = (const float*)d_in[5];
  const float* qmb  = (const float*)d_in[6];
  const float* pw   = (const float*)d_in[7];
  const float* pb   = (const float*)d_in[8];
  const float* n2w  = (const float*)d_in[9];
  const float* n2b  = (const float*)d_in[10];
  const float* f11w = (const float*)d_in[11];
  const float* f11b = (const float*)d_in[12];
  const float* f12w = (const float*)d_in[13];
  const float* f12b = (const float*)d_in[14];
  const float* f2w  = (const float*)d_in[15];
  const float* f2b  = (const float*)d_in[16];
  char* ws   = (char*)d_ws;
  float* out = (float*)d_out;

  _Float16* Wqkv  = (_Float16*)(ws + OFF_WQKV);
  float*    bqkv  = (float*)(ws + OFF_BQKV);
  _Float16* Wproj = (_Float16*)(ws + OFF_WPROJ);
  float*    bproj = (float*)(ws + OFF_BPROJ);
  _Float16* Wf11  = (_Float16*)(ws + OFF_WFC11);
  float*    bf11  = (float*)(ws + OFF_BFC11);
  _Float16* Wf12  = (_Float16*)(ws + OFF_WFC12);
  float*    bf12  = (float*)(ws + OFF_BFC12);
  _Float16* Wf2   = (_Float16*)(ws + OFF_WFC2);
  float*    bf2   = (float*)(ws + OFF_BFC2);
  _Float16* xoutp = (_Float16*)(ws + OFF_XOUT);

  // allow >64KB dynamic LDS (deterministic, no stream ops -> graph-capture safe)
  (void)hipFuncSetAttribute((const void*)k_qkv_attn,
                            hipFuncAttributeMaxDynamicSharedMemorySize, 229376);
  (void)hipFuncSetAttribute((const void*)k_mlp,
                            hipFuncAttributeMaxDynamicSharedMemorySize, 196608);

  k_prep<<<64, 256, 0, stream>>>(qsw, qsb, qmw, qmb, pw, pb,
                                 f11w, f11b, f12w, f12b, f2w, f2b, ws);
  k_qkv_attn<<<NWIN, 256, 229376, stream>>>(x, n1w, n1b, Wqkv, bqkv, xoutp);
  k_proj<<<NWIN, 256, 0, stream>>>(xoutp, Wproj, bproj, x, out);
  k_mlp<<<NWIN, 256, 196608, stream>>>(n2w, n2b, Wf11, bf11, Wf12, bf12, Wf2, bf2, out);
}
